// ServerGPT2Attention_36885179138282
// MI455X (gfx1250) — matrix operand order repaired and checked by reading, not compiled, not run on hardware
//
#include <hip/hip_runtime.h>
#include <hip/hip_bf16.h>

// MI455X / gfx1250 wave32 WMMA GPT2 attention block.
// Pipeline: [convert hidden->f16, weights->f16 transposed [N][K]] ->
//           [qkv GEMM + bias + lora, f16 head-split] -> [causal flash attention]
//           -> [proj GEMM + bias, fp32 out].
// Tile staging uses GLOBAL_LOAD_ASYNC_TO_LDS_B128 (ASYNCcnt) when available.
// Workspace (d_ws, halves): q(4M) k(4M) v(4M) attn(4M) hF16(4M) wAT(3M) wPT(1M) = 48MB.

#define S_LEN 2048
#define HID   1024
#define D3    3072
#define NH    16
#define HD    64

typedef __attribute__((ext_vector_type(16))) _Float16 v16h;
typedef __attribute__((ext_vector_type(8)))  _Float16 v8h;
typedef __attribute__((ext_vector_type(4)))  _Float16 v4h;
typedef __attribute__((ext_vector_type(8)))  float    v8f;
typedef __attribute__((ext_vector_type(4)))  int      v4i;

union V16U { v16h v; v8h h[2]; };

__device__ __forceinline__ v8f wmma_f16(const V16U& a, const V16U& b, v8f c) {
    return __builtin_amdgcn_wmma_f32_16x16x32_f16(false, a.v, false, b.v,
                                                  (short)0, c, false, false);
}

// ---- async global->LDS copy (gfx1250 async path, ASYNCcnt), with fallback ----
#if defined(__has_builtin)
#  if __has_builtin(__builtin_amdgcn_global_load_async_to_lds_b128)
#    define ASYNC_CP 1
#  endif
#endif
#ifndef ASYNC_CP
#  define ASYNC_CP 0
#endif

typedef __attribute__((address_space(1))) v4i* gptr_v4i;
typedef __attribute__((address_space(3))) v4i* lptr_v4i;

__device__ __forceinline__ void cp_b128(void* lds, const void* gmem) {
#if ASYNC_CP
    __builtin_amdgcn_global_load_async_to_lds_b128(
        (gptr_v4i)gmem, (lptr_v4i)lds, 0, 0);
#else
    *(v8h*)lds = *(const v8h*)gmem;
#endif
}

__device__ __forceinline__ void async_wait() {
#if ASYNC_CP
    asm volatile("s_wait_asynccnt 0x0" ::: "memory");
#endif
}

// ---------------------------------------------------------------------------
// Kernel 0a: hidden (fp32) -> f16, contiguous.
// ---------------------------------------------------------------------------
__global__ __launch_bounds__(256) void hcvt_kernel(const float* __restrict__ src,
                                                   _Float16* __restrict__ dst, int n4) {
    int i = blockIdx.x * 256 + threadIdx.x;
    if (i < n4) {
        const float4 v = ((const float4*)src)[i];
        v4h o;
        o[0] = (_Float16)v.x; o[1] = (_Float16)v.y;
        o[2] = (_Float16)v.z; o[3] = (_Float16)v.w;
        ((v4h*)dst)[i] = o;
    }
}

// ---------------------------------------------------------------------------
// Kernel 0b: w [K][N] fp32 -> wT [N][K] f16 (B-fragment-friendly layout).
// ---------------------------------------------------------------------------
__global__ __launch_bounds__(256) void wcvt_kernel(const float* __restrict__ w,
                                                   _Float16* __restrict__ wT,
                                                   int K, int N) {
    size_t i = (size_t)blockIdx.x * 256 + threadIdx.x;
    if (i < (size_t)K * N) {
        int k = (int)(i / N), n = (int)(i % N);
        wT[(size_t)n * K + k] = (_Float16)w[i];
    }
}

// ---------------------------------------------------------------------------
// Kernel 1: qkv = hidden @ w_attn + b_attn + lora ; split heads, store f16.
// M=4096, N=3072, K=1024. Block tile 128x64, BK=32. 8 waves (4x2), each wave
// owns a 32x32 C region (2 A-frags x 2 B-frags -> 4 WMMA / K-step).
// ---------------------------------------------------------------------------
__global__ __launch_bounds__(256) void qkv_kernel(
    const _Float16* __restrict__ hF, const _Float16* __restrict__ wT,
    const float* __restrict__ b_attn, const float* __restrict__ lora,
    _Float16* __restrict__ qB, _Float16* __restrict__ kB, _Float16* __restrict__ vB)
{
    __shared__ alignas(16) _Float16 Al[128][40];  // [m][k]
    __shared__ alignas(16) _Float16 Bl[64][40];   // [n][k]

    const int tid  = threadIdx.x;
    const int lane = tid & 31;
    const int wid  = tid >> 5;
    const int wm   = wid & 3;        // 4 wave-rows of 32
    const int wn   = wid >> 2;       // 2 wave-cols of 32
    const int mBase = blockIdx.y * 128;
    const int nBase = blockIdx.x * 64;

    v8f acc[2][2];
#pragma unroll
    for (int u = 0; u < 2; ++u)
#pragma unroll
        for (int t = 0; t < 2; ++t)
#pragma unroll
            for (int r = 0; r < 8; ++r) acc[u][t][r] = 0.0f;

    for (int k0 = 0; k0 < HID; k0 += 32) {
#pragma unroll
        for (int e = 0; e < 2; ++e) {                        // A: 128x32 = 512 b128
            int c = e * 256 + tid;
            int r = c >> 2, d0 = (c & 3) * 8;
            cp_b128(&Al[r][d0], &hF[(size_t)(mBase + r) * HID + k0 + d0]);
        }
        {                                                    // B: 64x32 = 256 b128
            int r = tid >> 2, d0 = (tid & 3) * 8;
            cp_b128(&Bl[r][d0], &wT[(size_t)(nBase + r) * HID + k0 + d0]);
        }
        async_wait();
        __syncthreads();

        V16U a[2];
        const int ka = (lane & 16) ? 16 : 0;
#pragma unroll
        for (int u = 0; u < 2; ++u) {
            const int am = wm * 32 + u * 16 + (lane & 15);
            a[u].h[0] = *(const v8h*)&Al[am][ka];
            a[u].h[1] = *(const v8h*)&Al[am][ka + 8];
        }
        const int kb = (lane & 16) ? 16 : 0;
#pragma unroll
        for (int t = 0; t < 2; ++t) {
            V16U b;
            const int bn = wn * 32 + t * 16 + (lane & 15);
            b.h[0] = *(const v8h*)&Bl[bn][kb];
            b.h[1] = *(const v8h*)&Bl[bn][kb + 8];
#pragma unroll
            for (int u = 0; u < 2; ++u) acc[u][t] = wmma_f16(a[u], b, acc[u][t]);
        }
        __syncthreads();
    }

    // Epilogue: + bias + lora, route into Q/K/V head-split f16 buffers.
#pragma unroll
    for (int t = 0; t < 2; ++t) {
        const int col = nBase + wn * 32 + t * 16 + (lane & 15);
        const float bv = b_attn[col];
        const int sec = col / HID, rem = col % HID;
        const int h = rem >> 6, d = rem & 63;
        _Float16* __restrict__ dst = (sec == 0) ? qB : (sec == 1) ? kB : vB;
#pragma unroll
        for (int u = 0; u < 2; ++u) {
            const int rBase = mBase + wm * 32 + u * 16 + ((lane & 16) ? 8 : 0);
#pragma unroll
            for (int r = 0; r < 8; ++r) {
                const int row = rBase + r;
                const float v = acc[u][t][r] + bv + lora[(size_t)row * D3 + col];
                const int bb = row >> 11, s = row & (S_LEN - 1);
                dst[(((size_t)(bb * NH + h)) * S_LEN + s) * HD + d] = (_Float16)v;
            }
        }
    }
}

// ---------------------------------------------------------------------------
// Kernel 2: causal attention for one (b,h), 64-row q-tile, streaming 64-wide
// key tiles with online softmax. 8 waves, each owns 2 of 16 subtiles.
// ---------------------------------------------------------------------------
__global__ __launch_bounds__(256) void attn_kernel(
    const _Float16* __restrict__ qB, const _Float16* __restrict__ kB,
    const _Float16* __restrict__ vB, _Float16* __restrict__ aB)
{
    __shared__ alignas(16) _Float16 Qs[64][72];   // [q][d]
    __shared__ alignas(16) _Float16 Ks[64][72];   // [key][d]
    __shared__ alignas(16) _Float16 Vs[64][72];   // [d][key] (transposed)
    __shared__ alignas(16) _Float16 Ps[64][72];   // probs [q][key]
    __shared__ alignas(16) float    Ss[64][72];   // scores f32
    __shared__ float rowM[64], rowL[64], rowA[64];

    const int tid  = threadIdx.x;
    const int lane = tid & 31;
    const int wid  = tid >> 5;
    const int qt   = blockIdx.x;
    const int bh   = blockIdx.y;
    const int bIdx = bh >> 4, hIdx = bh & 15;
    const size_t headOff = (size_t)bh * S_LEN * HD;
    const int qBase = qt * 64;

    const int i0 = wid >> 2, j0 = wid & 3;        // subtiles wid and wid+8
    const int i1 = i0 + 2;

#pragma unroll
    for (int e = 0; e < 2; ++e) {                 // Q tile: 64x64 f16 = 512 b128
        int c = e * 256 + tid;
        int r = c >> 3, d0 = (c & 7) * 8;
        cp_b128(&Qs[r][d0], &qB[headOff + (size_t)(qBase + r) * HD + d0]);
    }
    if (tid < 64) { rowM[tid] = -__builtin_inff(); rowL[tid] = 0.0f; }

    v8f oAcc[2];
#pragma unroll
    for (int t = 0; t < 2; ++t)
#pragma unroll
        for (int r = 0; r < 8; ++r) oAcc[t][r] = 0.0f;

    async_wait();
    __syncthreads();

    for (int kt = 0; kt <= qt; ++kt) {
        const int kBase = kt * 64;
#pragma unroll
        for (int e = 0; e < 2; ++e) {                         // K tile async
            int c = e * 256 + tid;
            int r = c >> 3, d0 = (c & 7) * 8;
            cp_b128(&Ks[r][d0], &kB[headOff + (size_t)(kBase + r) * HD + d0]);
        }
        for (int c = tid; c < 64 * 64; c += 256) {            // V tile, transposed
            int r = c >> 6, d = c & 63;
            Vs[d][r] = vB[headOff + (size_t)(kBase + r) * HD + d];
        }
        async_wait();
        __syncthreads();

        // scores: S = Q @ K^T
#pragma unroll
        for (int t = 0; t < 2; ++t) {
            const int i = t ? i1 : i0;
            v8f s;
#pragma unroll
            for (int r = 0; r < 8; ++r) s[r] = 0.0f;
#pragma unroll
            for (int kc = 0; kc < 2; ++kc) {
                V16U a, b;
                const int am = i * 16 + (lane & 15);
                const int ka = kc * 32 + ((lane & 16) ? 16 : 0);
                a.h[0] = *(const v8h*)&Qs[am][ka];
                a.h[1] = *(const v8h*)&Qs[am][ka + 8];
                const int bn = j0 * 16 + (lane & 15);
                const int kb = kc * 32 + ((lane & 16) ? 16 : 0);
                b.h[0] = *(const v8h*)&Ks[bn][kb];
                b.h[1] = *(const v8h*)&Ks[bn][kb + 8];
                s = wmma_f16(a, b, s);
            }
            const int mOff = i * 16 + ((lane & 16) ? 8 : 0);
            const int nCol = j0 * 16 + (lane & 15);
#pragma unroll
            for (int r = 0; r < 8; ++r) Ss[mOff + r][nCol] = s[r];
        }
        __syncthreads();

        // online softmax: one thread per query row
        if (tid < 64) {
            const int r = tid, qa = qBase + r;
            const float mOld = rowM[r];
            float mNew = mOld;
            for (int c = 0; c < 64; ++c) {
                float sv = Ss[r][c] * 0.125f;
                if (kBase + c > qa) sv = -__builtin_inff();
                Ss[r][c] = sv;
                mNew = fmaxf(mNew, sv);
            }
            const float alpha = __expf(mOld - mNew);
            float l = rowL[r] * alpha;
            for (int c = 0; c < 64; ++c) {
                const float p = __expf(Ss[r][c] - mNew);
                l += p;
                Ps[r][c] = (_Float16)p;
            }
            rowM[r] = mNew; rowL[r] = l; rowA[r] = alpha;
        }
        __syncthreads();

        // O = O*alpha + P @ V
#pragma unroll
        for (int t = 0; t < 2; ++t) {
            const int i = t ? i1 : i0;
            const int mOff = i * 16 + ((lane & 16) ? 8 : 0);
#pragma unroll
            for (int r = 0; r < 8; ++r) oAcc[t][r] *= rowA[mOff + r];
#pragma unroll
            for (int kc = 0; kc < 2; ++kc) {
                V16U a, b;
                const int am = i * 16 + (lane & 15);
                const int ka = kc * 32 + ((lane & 16) ? 16 : 0);
                a.h[0] = *(const v8h*)&Ps[am][ka];
                a.h[1] = *(const v8h*)&Ps[am][ka + 8];
                const int bn = j0 * 16 + (lane & 15);
                const int kb = kc * 32 + ((lane & 16) ? 16 : 0);
                b.h[0] = *(const v8h*)&Vs[bn][kb];
                b.h[1] = *(const v8h*)&Vs[bn][kb + 8];
                oAcc[t] = wmma_f16(a, b, oAcc[t]);
            }
        }
        __syncthreads();
    }

    // normalize and store merged-head f16 output
#pragma unroll
    for (int t = 0; t < 2; ++t) {
        const int i = t ? i1 : i0;
        const int mOff = i * 16 + ((lane & 16) ? 8 : 0);
        const int dCol = j0 * 16 + (lane & 15);
#pragma unroll
        for (int r = 0; r < 8; ++r) {
            const int row = mOff + r;
            const float v = oAcc[t][r] / rowL[row];
            aB[((size_t)bIdx * S_LEN + qBase + row) * HID + hIdx * HD + dCol] = (_Float16)v;
        }
    }
}

// ---------------------------------------------------------------------------
// Kernel 3: out = attn(f16) @ w_proj + b_proj  (M=4096, N=1024, K=1024)
// Block tile 128x64, same wave layout as kernel 1.
// ---------------------------------------------------------------------------
__global__ __launch_bounds__(256) void proj_kernel(
    const _Float16* __restrict__ aB, const _Float16* __restrict__ wT,
    const float* __restrict__ b_proj, float* __restrict__ out)
{
    __shared__ alignas(16) _Float16 Al[128][40];
    __shared__ alignas(16) _Float16 Bl[64][40];

    const int tid  = threadIdx.x;
    const int lane = tid & 31;
    const int wid  = tid >> 5;
    const int wm   = wid & 3;
    const int wn   = wid >> 2;
    const int mBase = blockIdx.y * 128;
    const int nBase = blockIdx.x * 64;

    v8f acc[2][2];
#pragma unroll
    for (int u = 0; u < 2; ++u)
#pragma unroll
        for (int t = 0; t < 2; ++t)
#pragma unroll
            for (int r = 0; r < 8; ++r) acc[u][t][r] = 0.0f;

    for (int k0 = 0; k0 < HID; k0 += 32) {
#pragma unroll
        for (int e = 0; e < 2; ++e) {
            int c = e * 256 + tid;
            int r = c >> 2, d0 = (c & 3) * 8;
            cp_b128(&Al[r][d0], &aB[(size_t)(mBase + r) * HID + k0 + d0]);
        }
        {
            int r = tid >> 2, d0 = (tid & 3) * 8;
            cp_b128(&Bl[r][d0], &wT[(size_t)(nBase + r) * HID + k0 + d0]);
        }
        async_wait();
        __syncthreads();

        V16U a[2];
        const int ka = (lane & 16) ? 16 : 0;
#pragma unroll
        for (int u = 0; u < 2; ++u) {
            const int am = wm * 32 + u * 16 + (lane & 15);
            a[u].h[0] = *(const v8h*)&Al[am][ka];
            a[u].h[1] = *(const v8h*)&Al[am][ka + 8];
        }
        const int kb = (lane & 16) ? 16 : 0;
#pragma unroll
        for (int t = 0; t < 2; ++t) {
            V16U b;
            const int bn = wn * 32 + t * 16 + (lane & 15);
            b.h[0] = *(const v8h*)&Bl[bn][kb];
            b.h[1] = *(const v8h*)&Bl[bn][kb + 8];
#pragma unroll
            for (int u = 0; u < 2; ++u) acc[u][t] = wmma_f16(a[u], b, acc[u][t]);
        }
        __syncthreads();
    }

#pragma unroll
    for (int t = 0; t < 2; ++t) {
        const int col = nBase + wn * 32 + t * 16 + (lane & 15);
        const float bv = b_proj[col];
#pragma unroll
        for (int u = 0; u < 2; ++u) {
            const int rBase = mBase + wm * 32 + u * 16 + ((lane & 16) ? 8 : 0);
#pragma unroll
            for (int r = 0; r < 8; ++r)
                out[(size_t)(rBase + r) * HID + col] = acc[u][t][r] + bv;
        }
    }
}

// ---------------------------------------------------------------------------
extern "C" void kernel_launch(void* const* d_in, const int* in_sizes, int n_in,
                              void* d_out, int out_size, void* d_ws, size_t ws_size,
                              hipStream_t stream) {
    (void)in_sizes; (void)n_in; (void)out_size; (void)ws_size;
    const float* hidden = (const float*)d_in[0];
    const float* lora   = (const float*)d_in[1];
    const float* w_attn = (const float*)d_in[2];
    const float* b_attn = (const float*)d_in[3];
    const float* w_proj = (const float*)d_in[4];
    const float* b_proj = (const float*)d_in[5];
    float* out = (float*)d_out;

    const size_t headElems = (size_t)2 * NH * S_LEN * HD;   // 4,194,304
    _Float16* qB  = (_Float16*)d_ws;
    _Float16* kB  = qB  + headElems;
    _Float16* vB  = kB  + headElems;
    _Float16* aB  = vB  + headElems;                 // merged attn [4096][1024]
    _Float16* hF  = aB  + headElems;                 // hidden f16 [4096][1024]
    _Float16* wAT = hF  + headElems;                 // w_attn^T f16 [3072][1024]
    _Float16* wPT = wAT + (size_t)D3 * HID;          // w_proj^T f16 [1024][1024]

    hcvt_kernel<<<(2 * S_LEN * HID / 4 + 255) / 256, 256, 0, stream>>>(
        hidden, hF, 2 * S_LEN * HID / 4);
    wcvt_kernel<<<((size_t)HID * D3 + 255) / 256, 256, 0, stream>>>(w_attn, wAT, HID, D3);
    wcvt_kernel<<<((size_t)HID * HID + 255) / 256, 256, 0, stream>>>(w_proj, wPT, HID, HID);

    qkv_kernel <<<dim3(D3 / 64, (2 * S_LEN) / 128), 256, 0, stream>>>(
        hF, wAT, b_attn, lora, qB, kB, vB);
    attn_kernel<<<dim3(S_LEN / 64, 2 * NH), 256, 0, stream>>>(qB, kB, vB, aB);
    proj_kernel<<<dim3(HID / 64, (2 * S_LEN) / 128), 256, 0, stream>>>(
        aB, wPT, b_proj, out);
}
